// VAE_20478404067351
// MI455X (gfx1250) — matrix operand order repaired and checked by reading, not compiled, not run on hardware
//
#include <hip/hip_runtime.h>
#include <hip/hip_bf16.h>
#include <math.h>

typedef __attribute__((ext_vector_type(16))) _Float16 v16h;
typedef __attribute__((ext_vector_type(8)))  float    v8f;

#define NEG_BIG (-3.402823466e+38f)

// ---------------------------------------------------------------------------
// Weight pre-pack: f32 [Cin][Cout] -> f16 fragments in B-register order.
// B (f16 32x16) layout per CDNA5 ISA: lane L holds col N=L&15;
// lanes 0-15: K = kt*32 + {0..7,16..23}, lanes 16-31: K = kt*32 + {8..15,24..31}.
// Packed layout: out[((kt*NT + nt)*32 + lane)*16 + j], zero-padded for k>=Cin.
// Launch: grid = KT*NT blocks, 32 threads.
// ---------------------------------------------------------------------------
__global__ void pack_w_kernel(const float* __restrict__ W, int Cin, int Cout,
                              _Float16* __restrict__ out)
{
  const int lane = threadIdx.x;            // 0..31
  const int tile = blockIdx.x;             // kt*NT + nt
  const int NT = Cout >> 4;
  const int nt = tile % NT, kt = tile / NT;
  const int col = nt * 16 + (lane & 15);
  const int kb  = kt * 32 + (lane >> 4) * 8;
  _Float16* o = out + ((size_t)tile * 32 + lane) * 16;
#pragma unroll
  for (int j = 0; j < 16; ++j) {
    int k = kb + j + ((j >> 3) << 3);
    o[j] = (_Float16)((k < Cin) ? W[(size_t)k * Cout + col] : 0.f);
  }
}

// ---------------------------------------------------------------------------
// WMMA MLP layer (compile-time shapes): out[16][COUT] = act(in @ W + bias).
// One wave per 16-row tile. in/out: per-wave LDS buffers, zero-padded to
// KT*32 input columns. Weights: pre-packed f16 fragments (see pack_w_kernel).
// A (f16 16x32): lane L holds row M=L&15; lanes 0-15: K in {0..7,16..23},
// lanes 16-31: K in {8..15,24..31} (+kt*32). C/D: row=(L>>4)*8+v, col=L&15.
// ---------------------------------------------------------------------------
template<int KT, int NT, int ISTR, int OSTR, bool RELU, bool HASB>
__device__ __forceinline__ void mlp_layer(const _Float16* __restrict__ wp,
                                          const float* __restrict__ Bv,
                                          const float* __restrict__ in,
                                          float* __restrict__ out, int lane)
{
  const int hl = lane >> 4;
  const int rn = lane & 15;
  v16h af[KT];
#pragma unroll
  for (int kt = 0; kt < KT; ++kt) {
    const float* s = in + rn * ISTR + kt * 32 + hl * 8;
    float4 p0 = *(const float4*)(s);
    float4 p1 = *(const float4*)(s + 4);
    float4 p2 = *(const float4*)(s + 16);
    float4 p3 = *(const float4*)(s + 20);
    af[kt][0]  = (_Float16)p0.x; af[kt][1]  = (_Float16)p0.y;
    af[kt][2]  = (_Float16)p0.z; af[kt][3]  = (_Float16)p0.w;
    af[kt][4]  = (_Float16)p1.x; af[kt][5]  = (_Float16)p1.y;
    af[kt][6]  = (_Float16)p1.z; af[kt][7]  = (_Float16)p1.w;
    af[kt][8]  = (_Float16)p2.x; af[kt][9]  = (_Float16)p2.y;
    af[kt][10] = (_Float16)p2.z; af[kt][11] = (_Float16)p2.w;
    af[kt][12] = (_Float16)p3.x; af[kt][13] = (_Float16)p3.y;
    af[kt][14] = (_Float16)p3.z; af[kt][15] = (_Float16)p3.w;
  }
  for (int nt = 0; nt < NT; ++nt) {
    v8f acc = {0.f, 0.f, 0.f, 0.f, 0.f, 0.f, 0.f, 0.f};
#pragma unroll
    for (int kt = 0; kt < KT; ++kt) {
      v16h bf = *(const v16h*)(wp + ((size_t)(kt * NT + nt) * 32 + lane) * 16);
      acc = __builtin_amdgcn_wmma_f32_16x16x32_f16(false, af[kt], false, bf,
                                                   (short)0, acc, false, false);
    }
    const int col = nt * 16 + rn;
    float bias = 0.f;
    if (HASB) bias = Bv[col];
#pragma unroll
    for (int v = 0; v < 8; ++v) {
      float x = acc[v] + bias;
      if (RELU) x = fmaxf(x, 0.f);
      out[(hl * 8 + v) * OSTR + col] = x;
    }
  }
}

// ---------------------------------------------------------------------------
// Farthest point sampling. One block per batch. Tie-break: first occurrence.
// ---------------------------------------------------------------------------
__global__ void fps_kernel(const float* __restrict__ pos, int N, int M,
                           int* __restrict__ idxOut, float* __restrict__ centersOut)
{
  __shared__ float sd[4096];
  __shared__ float rv[256];
  __shared__ int   ri[256];
  const int b = blockIdx.x, tid = threadIdx.x, T = blockDim.x;
  const float* p = pos + (size_t)b * N * 3;
  const float x0 = p[0], y0 = p[1], z0 = p[2];
  for (int i = tid; i < N; i += T) {
    float dx = p[3 * i] - x0, dy = p[3 * i + 1] - y0, dz = p[3 * i + 2] - z0;
    sd[i] = dx * dx + dy * dy + dz * dz;
  }
  if (tid == 0) {
    idxOut[(size_t)b * M] = 0;
    centersOut[(size_t)b * M * 3 + 0] = x0;
    centersOut[(size_t)b * M * 3 + 1] = y0;
    centersOut[(size_t)b * M * 3 + 2] = z0;
  }
  __syncthreads();
  for (int t = 1; t < M; ++t) {
    float bv = -1.f; int bi = 0;
    for (int i = tid; i < N; i += T) {
      float v = sd[i];
      if (v > bv) { bv = v; bi = i; }
    }
    rv[tid] = bv; ri[tid] = bi;
    __syncthreads();
    for (int s = T >> 1; s > 0; s >>= 1) {
      if (tid < s) {
        float v2 = rv[tid + s]; int i2 = ri[tid + s];
        if (v2 > rv[tid] || (v2 == rv[tid] && i2 < ri[tid])) { rv[tid] = v2; ri[tid] = i2; }
      }
      __syncthreads();
    }
    const int c = ri[0];
    const float cx = p[3 * c], cy = p[3 * c + 1], cz = p[3 * c + 2];
    if (tid == 0) {
      idxOut[(size_t)b * M + t] = c;
      centersOut[((size_t)b * M + t) * 3 + 0] = cx;
      centersOut[((size_t)b * M + t) * 3 + 1] = cy;
      centersOut[((size_t)b * M + t) * 3 + 2] = cz;
    }
    for (int i = tid; i < N; i += T) {
      float dx = p[3 * i] - cx, dy = p[3 * i + 1] - cy, dz = p[3 * i + 2] - cz;
      sd[i] = fminf(sd[i], dx * dx + dy * dy + dz * dz);
    }
    __syncthreads();
  }
}

// ---------------------------------------------------------------------------
// K(=64)-nearest within radius. One thread per center; 64-entry max-heap.
// ---------------------------------------------------------------------------
__global__ void radius_kernel(const float* __restrict__ pos, const float* __restrict__ centers,
                              int N, int M, float r2,
                              int* __restrict__ nbOut, int* __restrict__ cntOut)
{
  const int g = blockIdx.x * blockDim.x + threadIdx.x;   // over B*M centers
  const int b = g / M;
  const float cx = centers[(size_t)g * 3], cy = centers[(size_t)g * 3 + 1],
              cz = centers[(size_t)g * 3 + 2];
  const float* p = pos + (size_t)b * N * 3;
  float hd[64]; int hi[64]; int cnt = 0;
  for (int i = 0; i < N; ++i) {
    float dx = p[3 * i] - cx, dy = p[3 * i + 1] - cy, dz = p[3 * i + 2] - cz;
    float d = dx * dx + dy * dy + dz * dz;
    if (d <= r2) {
      if (cnt < 64) {
        int j = cnt++;
        hd[j] = d; hi[j] = i;
        while (j > 0) {
          int par = (j - 1) >> 1;
          if (hd[par] < hd[j]) {
            float td = hd[par]; hd[par] = hd[j]; hd[j] = td;
            int ti = hi[par]; hi[par] = hi[j]; hi[j] = ti;
            j = par;
          } else break;
        }
      } else if (d < hd[0]) {
        hd[0] = d; hi[0] = i;
        int j = 0;
        for (;;) {
          int l = 2 * j + 1, r = 2 * j + 2, big = j;
          if (l < 64 && hd[l] > hd[big]) big = l;
          if (r < 64 && hd[r] > hd[big]) big = r;
          if (big == j) break;
          float td = hd[big]; hd[big] = hd[j]; hd[j] = td;
          int ti = hi[big]; hi[big] = hi[j]; hi[j] = ti;
          j = big;
        }
      }
    }
  }
  for (int k = 0; k < 64; ++k) nbOut[(size_t)g * 64 + k] = (k < cnt) ? hi[k] : 0;
  cntOut[g] = cnt;
}

// ---------------------------------------------------------------------------
// SA1: rel(3) -> 32 relu -> 32 relu -> 64 (+bias), masked max over K=64.
// ---------------------------------------------------------------------------
__global__ void sa1_kernel(const float* __restrict__ pos, const float* __restrict__ centers,
                           const int* __restrict__ nb, const int* __restrict__ cnts,
                           const _Float16* pw1a, const float* b1a,
                           const _Float16* pw1b, const float* b1b,
                           const _Float16* pw1c, const float* b1c,
                           float* __restrict__ xout)
{
  __shared__ alignas(16) float bufA[4][16 * 64];
  __shared__ alignas(16) float bufB[4][16 * 64];
  const int wid = threadIdx.x >> 5, lane = threadIdx.x & 31;
  const int cidx = blockIdx.x * 4 + wid;
  const int b = cidx >> 11;
  const float cx = centers[(size_t)cidx * 3], cy = centers[(size_t)cidx * 3 + 1],
              cz = centers[(size_t)cidx * 3 + 2];
  const int cnt = cnts[cidx];
  const int* nbp = nb + (size_t)cidx * 64;
  float* A = bufA[wid]; float* Bb = bufB[wid];
  const int hl = lane >> 4, row = lane & 15;
  float m0 = NEG_BIG, m1 = NEG_BIG;
  for (int t = 0; t < 4; ++t) {
    int idx = nbp[t * 16 + row];   // invalid slots hold 0; excluded from max
    const float* pp = pos + ((size_t)b * 4096 + idx) * 3;
    float rx = pp[0] - cx, ry = pp[1] - cy, rz = pp[2] - cz;
#pragma unroll
    for (int j = 0; j < 16; ++j) {
      int c = hl * 16 + j;
      A[row * 64 + c] = (c == 0) ? rx : ((c == 1) ? ry : ((c == 2) ? rz : 0.f));
    }
    __syncthreads();
    mlp_layer<1, 2, 64, 64, true,  true>(pw1a, b1a, A,  Bb, lane);
    __syncthreads();
    mlp_layer<1, 2, 64, 64, true,  true>(pw1b, b1b, Bb, A,  lane);
    __syncthreads();
    mlp_layer<1, 4, 64, 64, false, true>(pw1c, b1c, A,  Bb, lane);
    __syncthreads();
    int lim = cnt - t * 16;
    for (int r = 0; r < 16; ++r) {
      if (r < lim) {
        m0 = fmaxf(m0, Bb[r * 64 + lane]);
        m1 = fmaxf(m1, Bb[r * 64 + lane + 32]);
      }
    }
    __syncthreads();
  }
  xout[(size_t)cidx * 64 + lane]      = m0;
  xout[(size_t)cidx * 64 + lane + 32] = m1;
}

// ---------------------------------------------------------------------------
// SA2: [x1(64) | rel(3)] (67, pad 96) -> 64 relu -> 64 relu -> 128, max.
// ---------------------------------------------------------------------------
__global__ void sa2_kernel(const float* __restrict__ pos1, const float* __restrict__ x1,
                           const float* __restrict__ centers, const int* __restrict__ nb,
                           const int* __restrict__ cnts,
                           const _Float16* pw2a, const float* b2a,
                           const _Float16* pw2b, const float* b2b,
                           const _Float16* pw2c, const float* b2c,
                           float* __restrict__ xout)
{
  __shared__ alignas(16) float bufA[2][16 * 128];
  __shared__ alignas(16) float bufB[2][16 * 128];
  const int wid = threadIdx.x >> 5, lane = threadIdx.x & 31;
  const int cidx = blockIdx.x * 2 + wid;
  const int b = cidx >> 9;
  const float cx = centers[(size_t)cidx * 3], cy = centers[(size_t)cidx * 3 + 1],
              cz = centers[(size_t)cidx * 3 + 2];
  const int cnt = cnts[cidx];
  const int* nbp = nb + (size_t)cidx * 64;
  float* A = bufA[wid]; float* Bb = bufB[wid];
  const int hl = lane >> 4, row = lane & 15;
  float mx[4] = {NEG_BIG, NEG_BIG, NEG_BIG, NEG_BIG};
  for (int t = 0; t < 4; ++t) {
    int idx = nbp[t * 16 + row];
    const float* xr = x1 + ((size_t)b * 2048 + idx) * 64 + hl * 32;
#pragma unroll
    for (int j = 0; j < 8; ++j)
      *(float4*)(A + row * 128 + hl * 32 + j * 4) = *(const float4*)(xr + j * 4);
    const float* pp = pos1 + ((size_t)b * 2048 + idx) * 3;
    float rx = pp[0] - cx, ry = pp[1] - cy, rz = pp[2] - cz;
#pragma unroll
    for (int j = 0; j < 16; ++j) {
      int c = 64 + hl * 16 + j;
      A[row * 128 + c] = (c == 64) ? rx : ((c == 65) ? ry : ((c == 66) ? rz : 0.f));
    }
    __syncthreads();
    mlp_layer<3, 4, 128, 128, true,  true>(pw2a, b2a, A,  Bb, lane);
    __syncthreads();
    mlp_layer<2, 4, 128, 128, true,  true>(pw2b, b2b, Bb, A,  lane);
    __syncthreads();
    mlp_layer<2, 8, 128, 128, false, true>(pw2c, b2c, A,  Bb, lane);
    __syncthreads();
    int lim = cnt - t * 16;
    for (int r = 0; r < 16; ++r) {
      if (r < lim) {
#pragma unroll
        for (int j = 0; j < 4; ++j) mx[j] = fmaxf(mx[j], Bb[r * 128 + lane + 32 * j]);
      }
    }
    __syncthreads();
  }
#pragma unroll
  for (int j = 0; j < 4; ++j) xout[(size_t)cidx * 128 + lane + 32 * j] = mx[j];
}

// ---------------------------------------------------------------------------
// g: [x2(128)|p2(3)] (131, pad 160) -> 128 relu -> 256 relu -> 512, tile max.
// ---------------------------------------------------------------------------
__global__ void g_kernel(const float* __restrict__ x2, const float* __restrict__ centers2,
                         const _Float16* pw3a, const float* b3a,
                         const _Float16* pw3b, const float* b3b,
                         const _Float16* pw3c, const float* b3c,
                         float* __restrict__ gpart)
{
  __shared__ alignas(16) float A[16 * 256];
  __shared__ alignas(16) float Bb[16 * 512];
  const int lane = threadIdx.x & 31;
  const int hl = lane >> 4, row = lane & 15;
  const int rowbase = blockIdx.x * 16;
  const int gr = rowbase + row;
  const float* xrow = x2 + (size_t)gr * 128 + hl * 64;
#pragma unroll
  for (int j = 0; j < 16; ++j)
    *(float4*)(A + row * 256 + hl * 64 + j * 4) = *(const float4*)(xrow + j * 4);
  const float* pr = centers2 + (size_t)gr * 3;
  float px = pr[0], py = pr[1], pz = pr[2];
#pragma unroll
  for (int j = 0; j < 16; ++j) {
    int c = 128 + hl * 16 + j;
    A[row * 256 + c] = (c == 128) ? px : ((c == 129) ? py : ((c == 130) ? pz : 0.f));
  }
  __syncthreads();
  mlp_layer<5, 8,  256, 512, true,  true>(pw3a, b3a, A,  Bb, lane);
  __syncthreads();
  mlp_layer<4, 16, 512, 256, true,  true>(pw3b, b3b, Bb, A,  lane);
  __syncthreads();
  mlp_layer<8, 32, 256, 512, false, true>(pw3c, b3c, A,  Bb, lane);
  __syncthreads();
  for (int j = 0; j < 16; ++j) {
    int c = lane + 32 * j;
    float m = Bb[c];
    for (int r = 1; r < 16; ++r) m = fmaxf(m, Bb[r * 512 + c]);
    gpart[(size_t)blockIdx.x * 512 + c] = m;
  }
}

__global__ void gmax_kernel(const float* __restrict__ gpart, float* __restrict__ feat)
{
  const int b = blockIdx.x;
  for (int c = threadIdx.x; c < 512; c += blockDim.x) {
    float m = NEG_BIG;
    for (int t = 0; t < 32; ++t) m = fmaxf(m, gpart[(size_t)(b * 32 + t) * 512 + c]);
    feat[(size_t)b * 512 + c] = m;
  }
}

__global__ void latent_kernel(const float* __restrict__ feat,
                              const float* __restrict__ wmu, const float* __restrict__ bmu,
                              const float* __restrict__ wlv, const float* __restrict__ blv,
                              const float* __restrict__ eps,
                              float* __restrict__ mu_out, float* __restrict__ lv_out,
                              float* __restrict__ z_out)
{
  const int t = threadIdx.x;
  const int b = t >> 5, j = t & 31;
  const float* f = feat + (size_t)b * 512;
  float smu = bmu[j], slv = blv[j];
  for (int k = 0; k < 512; ++k) {
    float fv = f[k];
    smu += fv * wmu[(size_t)k * 32 + j];
    slv += fv * wlv[(size_t)k * 32 + j];
  }
  mu_out[t] = smu;
  lv_out[t] = slv;
  z_out[t]  = smu + eps[t] * expf(0.5f * slv);
}

// ---------------------------------------------------------------------------
// Decoder: [q,sin q,cos q (9) | z(32)] (41, pad 64) -> 64 relu x4 -> dot d4.
// ---------------------------------------------------------------------------
__global__ void dec_kernel(const float* __restrict__ qp, const float* __restrict__ z,
                           const _Float16* pd0, const _Float16* pd1,
                           const _Float16* pd2, const _Float16* pd3,
                           const float* __restrict__ d4, float* __restrict__ occ)
{
  __shared__ alignas(16) float bufA[4][16 * 64];
  __shared__ alignas(16) float bufB[4][16 * 64];
  const int wid = threadIdx.x >> 5, lane = threadIdx.x & 31;
  const int tile = blockIdx.x * 4 + wid;
  const int rowbase = tile * 16;
  const int b = rowbase >> 13;
  float* A = bufA[wid]; float* Bb = bufB[wid];
  const int hl = lane >> 4, row = lane & 15;
  const int gr = rowbase + row;
  const float* qr = qp + (size_t)gr * 3;
  const float* zz = z + b * 32;
  float qx = qr[0], qy = qr[1], qz = qr[2];
  if (hl == 0) {
    float emb[9];
    emb[0] = qx; emb[1] = qy; emb[2] = qz;
    emb[3] = sinf(qx); emb[4] = sinf(qy); emb[5] = sinf(qz);
    emb[6] = cosf(qx); emb[7] = cosf(qy); emb[8] = cosf(qz);
#pragma unroll
    for (int j = 0; j < 32; ++j)
      A[row * 64 + j] = (j < 9) ? emb[j] : zz[j - 9];
  } else {
#pragma unroll
    for (int j = 0; j < 32; ++j) {
      int c = 32 + j;
      A[row * 64 + c] = (c < 41) ? zz[c - 9] : 0.f;
    }
  }
  __syncthreads();
  mlp_layer<2, 4, 64, 64, true, false>(pd0, nullptr, A,  Bb, lane);
  __syncthreads();
  mlp_layer<2, 4, 64, 64, true, false>(pd1, nullptr, Bb, A,  lane);
  __syncthreads();
  mlp_layer<2, 4, 64, 64, true, false>(pd2, nullptr, A,  Bb, lane);
  __syncthreads();
  mlp_layer<2, 4, 64, 64, true, false>(pd3, nullptr, Bb, A,  lane);
  __syncthreads();
  if (lane < 16) {
    float s = 0.f;
    const float* ar = A + lane * 64;
#pragma unroll
    for (int k = 0; k < 16; ++k) {
      float4 av = *(const float4*)(ar + k * 4);
      float4 wv = *(const float4*)(d4 + k * 4);
      s += av.x * wv.x + av.y * wv.y + av.z * wv.z + av.w * wv.w;
    }
    occ[rowbase + lane] = s;
  }
}

// ---------------------------------------------------------------------------
extern "C" void kernel_launch(void* const* d_in, const int* in_sizes, int n_in,
                              void* d_out, int out_size, void* d_ws, size_t ws_size,
                              hipStream_t stream)
{
  (void)in_sizes; (void)n_in; (void)out_size; (void)ws_size;
  const float* pos = (const float*)d_in[0];
  const float* qp  = (const float*)d_in[1];
  const float* eps = (const float*)d_in[2];
  const float* w1a = (const float*)d_in[3],  *w1b = (const float*)d_in[4],  *w1c = (const float*)d_in[5];
  const float* w2a = (const float*)d_in[6],  *w2b = (const float*)d_in[7],  *w2c = (const float*)d_in[8];
  const float* w3a = (const float*)d_in[9],  *w3b = (const float*)d_in[10], *w3c = (const float*)d_in[11];
  const float* wmu = (const float*)d_in[12], *wlv = (const float*)d_in[13];
  const float* dw0 = (const float*)d_in[14], *dw1 = (const float*)d_in[15], *dw2 = (const float*)d_in[16];
  const float* dw3 = (const float*)d_in[17], *dw4 = (const float*)d_in[18];
  const float* b1a = (const float*)d_in[19], *b1b = (const float*)d_in[20], *b1c = (const float*)d_in[21];
  const float* b2a = (const float*)d_in[22], *b2b = (const float*)d_in[23], *b2c = (const float*)d_in[24];
  const float* b3a = (const float*)d_in[25], *b3b = (const float*)d_in[26], *b3c = (const float*)d_in[27];
  const float* bmu = (const float*)d_in[28], *blv = (const float*)d_in[29];

  char* w = (char*)d_ws;
  size_t off = 0;
  auto carve = [&](size_t bytes) {
    char* p = w + off;
    off = (off + bytes + 255) & ~(size_t)255;
    return p;
  };
  int*   idx1  = (int*)  carve((size_t)8 * 2048 * 4);
  float* c1    = (float*)carve((size_t)8 * 2048 * 3 * 4);
  int*   nb1   = (int*)  carve((size_t)8 * 2048 * 64 * 4);
  int*   cnt1  = (int*)  carve((size_t)8 * 2048 * 4);
  float* x1    = (float*)carve((size_t)8 * 2048 * 64 * 4);
  int*   idx2  = (int*)  carve((size_t)8 * 512 * 4);
  float* c2    = (float*)carve((size_t)8 * 512 * 3 * 4);
  int*   nb2   = (int*)  carve((size_t)8 * 512 * 64 * 4);
  int*   cnt2  = (int*)  carve((size_t)8 * 512 * 4);
  float* x2    = (float*)carve((size_t)8 * 512 * 128 * 4);
  float* gpart = (float*)carve((size_t)256 * 512 * 4);
  float* feat  = (float*)carve((size_t)8 * 512 * 4);
  float* zl    = (float*)carve((size_t)8 * 32 * 4);
  // packed f16 weights: KT*NT*1024 bytes each
  _Float16* pw1a = (_Float16*)carve(1 * 2  * 1024);
  _Float16* pw1b = (_Float16*)carve(1 * 2  * 1024);
  _Float16* pw1c = (_Float16*)carve(1 * 4  * 1024);
  _Float16* pw2a = (_Float16*)carve(3 * 4  * 1024);
  _Float16* pw2b = (_Float16*)carve(2 * 4  * 1024);
  _Float16* pw2c = (_Float16*)carve(2 * 8  * 1024);
  _Float16* pw3a = (_Float16*)carve(5 * 8  * 1024);
  _Float16* pw3b = (_Float16*)carve(4 * 16 * 1024);
  _Float16* pw3c = (_Float16*)carve(8 * 32 * 1024);
  _Float16* pd0  = (_Float16*)carve(2 * 4  * 1024);
  _Float16* pd1  = (_Float16*)carve(2 * 4  * 1024);
  _Float16* pd2  = (_Float16*)carve(2 * 4  * 1024);
  _Float16* pd3  = (_Float16*)carve(2 * 4  * 1024);

  float* out    = (float*)d_out;
  float* mu_out = out + 65536;
  float* lv_out = out + 65536 + 256;

  hipLaunchKernelGGL(pack_w_kernel, dim3(1 * 2),  dim3(32), 0, stream, w1a, 3,   32,  pw1a);
  hipLaunchKernelGGL(pack_w_kernel, dim3(1 * 2),  dim3(32), 0, stream, w1b, 32,  32,  pw1b);
  hipLaunchKernelGGL(pack_w_kernel, dim3(1 * 4),  dim3(32), 0, stream, w1c, 32,  64,  pw1c);
  hipLaunchKernelGGL(pack_w_kernel, dim3(3 * 4),  dim3(32), 0, stream, w2a, 67,  64,  pw2a);
  hipLaunchKernelGGL(pack_w_kernel, dim3(2 * 4),  dim3(32), 0, stream, w2b, 64,  64,  pw2b);
  hipLaunchKernelGGL(pack_w_kernel, dim3(2 * 8),  dim3(32), 0, stream, w2c, 64,  128, pw2c);
  hipLaunchKernelGGL(pack_w_kernel, dim3(5 * 8),  dim3(32), 0, stream, w3a, 131, 128, pw3a);
  hipLaunchKernelGGL(pack_w_kernel, dim3(4 * 16), dim3(32), 0, stream, w3b, 128, 256, pw3b);
  hipLaunchKernelGGL(pack_w_kernel, dim3(8 * 32), dim3(32), 0, stream, w3c, 256, 512, pw3c);
  hipLaunchKernelGGL(pack_w_kernel, dim3(2 * 4),  dim3(32), 0, stream, dw0, 41,  64,  pd0);
  hipLaunchKernelGGL(pack_w_kernel, dim3(2 * 4),  dim3(32), 0, stream, dw1, 64,  64,  pd1);
  hipLaunchKernelGGL(pack_w_kernel, dim3(2 * 4),  dim3(32), 0, stream, dw2, 64,  64,  pd2);
  hipLaunchKernelGGL(pack_w_kernel, dim3(2 * 4),  dim3(32), 0, stream, dw3, 64,  64,  pd3);

  hipLaunchKernelGGL(fps_kernel,    dim3(8),    dim3(256), 0, stream, pos, 4096, 2048, idx1, c1);
  hipLaunchKernelGGL(radius_kernel, dim3(64),   dim3(256), 0, stream, pos, c1, 4096, 2048, 0.04f, nb1, cnt1);
  hipLaunchKernelGGL(sa1_kernel,    dim3(4096), dim3(128), 0, stream, pos, c1, nb1, cnt1,
                     pw1a, b1a, pw1b, b1b, pw1c, b1c, x1);
  hipLaunchKernelGGL(fps_kernel,    dim3(8),    dim3(256), 0, stream, c1, 2048, 512, idx2, c2);
  hipLaunchKernelGGL(radius_kernel, dim3(16),   dim3(256), 0, stream, c1, c2, 2048, 512, 0.16f, nb2, cnt2);
  hipLaunchKernelGGL(sa2_kernel,    dim3(2048), dim3(64),  0, stream, c1, x1, c2, nb2, cnt2,
                     pw2a, b2a, pw2b, b2b, pw2c, b2c, x2);
  hipLaunchKernelGGL(g_kernel,      dim3(256),  dim3(32),  0, stream, x2, c2,
                     pw3a, b3a, pw3b, b3b, pw3c, b3c, gpart);
  hipLaunchKernelGGL(gmax_kernel,   dim3(8),    dim3(256), 0, stream, gpart, feat);
  hipLaunchKernelGGL(latent_kernel, dim3(1),    dim3(256), 0, stream, feat, wmu, bmu, wlv, blv, eps,
                     mu_out, lv_out, zl);
  hipLaunchKernelGGL(dec_kernel,    dim3(1024), dim3(128), 0, stream, qp, zl,
                     pd0, pd1, pd2, pd3, dw4, out);
}
